// DistributionalCritic_65017214926887
// MI455X (gfx1250) — matrix operand order repaired and checked by reading, not compiled, not run on hardware
//
#include <hip/hip_runtime.h>
#include <hip/hip_bf16.h>
#include <stdint.h>

// ---------------------------------------------------------------------------
// Shapes (compile-time constants from the reference)
// ---------------------------------------------------------------------------
constexpr int kB  = 16384;
constexpr int kSD = 512;
constexpr int kAD = 64;
constexpr int kH  = 1024;
constexpr int kQE = 64;
constexpr int kNQ = 64;
constexpr int kKIN = kSD + kAD;       // 576

constexpr int kLDK = 40;              // padded LDS row stride in halfs (32 + 8 pad)

typedef __attribute__((ext_vector_type(16))) _Float16 v16h;
typedef __attribute__((ext_vector_type(8)))  _Float16 v8h;
typedef __attribute__((ext_vector_type(4)))  _Float16 v4h;
typedef __attribute__((ext_vector_type(8)))  float    v8f;

// ---------------------------------------------------------------------------
// gfx1250 async global->LDS copy (ASYNCcnt-tracked), inline asm per
// CDNA5 ISA (VGLOBAL GVS mode: saddr base + 32-bit voffset).
// LDS byte offset = low 32 bits of the generic pointer (flat LDS aperture keeps
// the LDS address in addr[31:0]).
// ---------------------------------------------------------------------------
__device__ __forceinline__ uint32_t lds_off_of(const void* p) {
    return (uint32_t)(uintptr_t)p;
}
__device__ __forceinline__ void async_load_b128(uint32_t lds_off, const void* base,
                                                uint32_t voff) {
    asm volatile("global_load_async_to_lds_b128 %0, %1, %2"
                 :: "v"(lds_off), "v"(voff), "s"(base) : "memory");
}
template <int N>
__device__ __forceinline__ void wait_async() {
    asm volatile("s_wait_asynccnt %0" :: "i"(N) : "memory");
}

// ---------------------------------------------------------------------------
// One-shot fp32 -> f16 conversion kernels (prep; tiny vs GEMM cost)
// ---------------------------------------------------------------------------
__global__ __launch_bounds__(256) void
cvt_xcat_kernel(const float* __restrict__ state, const float* __restrict__ action,
                _Float16* __restrict__ xcat)
{
    const int r = blockIdx.x;
    const int t = threadIdx.x;
    if (t < 128) {
        const float4 f = *(const float4*)(state + (size_t)r * kSD + t * 4);
        v4h h = { (_Float16)f.x, (_Float16)f.y, (_Float16)f.z, (_Float16)f.w };
        *(v4h*)(xcat + (size_t)r * kKIN + t * 4) = h;
    } else if (t < 144) {
        const int i = t - 128;
        const float4 f = *(const float4*)(action + (size_t)r * kAD + i * 4);
        v4h h = { (_Float16)f.x, (_Float16)f.y, (_Float16)f.z, (_Float16)f.w };
        *(v4h*)(xcat + (size_t)r * kKIN + kSD + i * 4) = h;
    }
}

__global__ __launch_bounds__(256) void
cvt_f32_to_f16_kernel(const float* __restrict__ src, _Float16* __restrict__ dst, int n4)
{
    const int i = blockIdx.x * 256 + threadIdx.x;
    if (i < n4) {
        const float4 f = ((const float4*)src)[i];
        v4h h = { (_Float16)f.x, (_Float16)f.y, (_Float16)f.z, (_Float16)f.w };
        ((v4h*)dst)[i] = h;
    }
}

// ---------------------------------------------------------------------------
// Async 3-stage pipelined WMMA GEMM:
//   C[kB, kH] = A[kB, K] * W[kH, K]^T     (A, W f16 row-major; C fp32)
// Block = 256 threads (8 waves), block tile 256(M) x 128(N), BK = 32.
// Waves 4(M) x 2(N): each wave owns 64x64 = 4x4 wmma_f32_16x16x32_f16.
// Tiles staged with global_load_async_to_lds_b128 into 3 rotating LDS buffers.
// Each stage issues exactly 6 async b128 per thread; ASYNCcnt completes in
// order, so `s_wait_asynccnt 6` = "oldest tile landed, one still in flight"
// (2-deep prefetch).  Final iteration drains with `s_wait_asynccnt 0`.
// ---------------------------------------------------------------------------
template <int K>
__global__ __launch_bounds__(256) void
gemm_async_kernel(const _Float16* __restrict__ A, const _Float16* __restrict__ W,
                  float* __restrict__ C)
{
    __shared__ _Float16 At[3][256 * kLDK];
    __shared__ _Float16 Bt[3][128 * kLDK];

    const int tid  = threadIdx.x;
    const int lane = tid & 31;
    const int wave = tid >> 5;
    const int wm   = (wave >> 1) * 64;     // wave M origin in block tile
    const int wn   = (wave & 1) * 64;      // wave N origin in block tile
    const int m0   = blockIdx.y * 256;
    const int n0   = blockIdx.x * 128;

    const uint32_t atab[3] = { lds_off_of(&At[0][0]), lds_off_of(&At[1][0]),
                               lds_off_of(&At[2][0]) };
    const uint32_t btab[3] = { lds_off_of(&Bt[0][0]), lds_off_of(&Bt[1][0]),
                               lds_off_of(&Bt[2][0]) };

    // stage one BKx{256,128} pair of tiles into LDS buffer `buf` (6 asyncs/thread)
    auto stage = [&](int buf, int k0) {
        // A tile: 256 rows x 32 halfs = 1024 x 16B ; 4 transfers/thread
#pragma unroll
        for (int i = 0; i < 4; ++i) {
            const int e = i * 256 + tid;
            const int r = e >> 2;
            const int s = (e & 3) * 8;
            async_load_b128(atab[buf] + (uint32_t)(r * kLDK + s) * 2,
                            (const void*)A,
                            (uint32_t)((((size_t)(m0 + r)) * K + k0 + s) * 2));
        }
        // B tile: 128 rows x 32 halfs = 512 x 16B ; 2 transfers/thread
#pragma unroll
        for (int i = 0; i < 2; ++i) {
            const int e = i * 256 + tid;
            const int r = e >> 2;
            const int s = (e & 3) * 8;
            async_load_b128(btab[buf] + (uint32_t)(r * kLDK + s) * 2,
                            (const void*)W,
                            (uint32_t)((((size_t)(n0 + r)) * K + k0 + s) * 2));
        }
    };

    v8f acc[4][4];
#pragma unroll
    for (int i = 0; i < 4; ++i)
#pragma unroll
        for (int j = 0; j < 4; ++j)
#pragma unroll
            for (int e = 0; e < 8; ++e) acc[i][j][e] = 0.0f;

    const int rsel = lane & 15;
    const int hi   = lane >> 4;

    constexpr int KT = K / 32;
    stage(0, 0);
    if (KT > 1) stage(1, 32);

    for (int kt = 0; kt < KT; ++kt) {
        // tile kt landed in LDS (own slice); keep at most one tile in flight
        if (kt + 1 < KT) wait_async<6>();
        else             wait_async<0>();
        __syncthreads();   // everyone's slice landed; buf (kt+2)%3 fully read

        if (kt + 2 < KT) stage((kt + 2) % 3, (kt + 2) * 32);

        const _Float16* at = &At[kt % 3][0];
        const _Float16* bt = &Bt[kt % 3][0];

        // A 16x32 f16 fragment: lanes 0-15 (M=lane): halfs = K0..7 | K16..23
        //                       lanes16-31 (M=lane-16): halfs = K8..15 | K24..31
        v16h af[4];
#pragma unroll
        for (int mi = 0; mi < 4; ++mi) {
            const _Float16* ap = at + (wm + mi * 16 + rsel) * kLDK + hi * 8;
            v16h a;
            *(v8h*)&a         = *(const v8h*)ap;
            *(((v8h*)&a) + 1) = *(const v8h*)(ap + 16);
            af[mi] = a;
        }
        // B 32x16 f16 fragment: lanes 0-15 (N=lane): halfs = K0..7 | K16..23
        //                       lanes16-31: halfs = K8..15 | K24..31
        v16h bfg[4];
#pragma unroll
        for (int ni = 0; ni < 4; ++ni) {
            const _Float16* bp = bt + (wn + ni * 16 + rsel) * kLDK + hi * 8;
            v16h b;
            *(v8h*)&b         = *(const v8h*)bp;
            *(((v8h*)&b) + 1) = *(const v8h*)(bp + 16);
            bfg[ni] = b;
        }

#pragma unroll
        for (int mi = 0; mi < 4; ++mi)
#pragma unroll
            for (int ni = 0; ni < 4; ++ni)
                acc[mi][ni] = __builtin_amdgcn_wmma_f32_16x16x32_f16(
                    false, af[mi], false, bfg[ni], (short)0, acc[mi][ni], false, false);
    }

    // epilogue: D layout -> N = lane&15, M = vgpr + 8*(lane>=16)
#pragma unroll
    for (int mi = 0; mi < 4; ++mi)
#pragma unroll
        for (int ni = 0; ni < 4; ++ni) {
            const int col  = n0 + wn + ni * 16 + rsel;
            const int rowb = m0 + wm + mi * 16 + hi * 8;
#pragma unroll
            for (int v = 0; v < 8; ++v)
                C[(size_t)(rowb + v) * kH + col] = acc[mi][ni][v];
        }
}

// ---------------------------------------------------------------------------
// Fused bias + LayerNorm + ReLU + f32->f16.  One block (256 thr) per row.
// ---------------------------------------------------------------------------
__global__ __launch_bounds__(256) void
ln_relu_kernel(const float* __restrict__ C, const float* __restrict__ bias,
               const float* __restrict__ g, const float* __restrict__ beta,
               _Float16* __restrict__ out)
{
    __shared__ float r1[256];
    __shared__ float r2[256];
    const int row = blockIdx.x;
    const int tid = threadIdx.x;

    float v[4];
    float s1 = 0.f, s2 = 0.f;
#pragma unroll
    for (int i = 0; i < 4; ++i) {
        const int h = i * 256 + tid;
        const float x = C[(size_t)row * kH + h] + bias[h];
        v[i] = x; s1 += x; s2 += x * x;
    }
    r1[tid] = s1; r2[tid] = s2;
    __syncthreads();
    for (int off = 128; off > 0; off >>= 1) {
        if (tid < off) { r1[tid] += r1[tid + off]; r2[tid] += r2[tid + off]; }
        __syncthreads();
    }
    const float mu  = r1[0] * (1.0f / kH);
    const float var = r2[0] * (1.0f / kH) - mu * mu;
    const float rs  = rsqrtf(var + 1e-5f);
#pragma unroll
    for (int i = 0; i < 4; ++i) {
        const int h = i * 256 + tid;
        float y = (v[i] - mu) * rs * g[h] + beta[h];
        y = fmaxf(y, 0.0f);
        out[(size_t)row * kH + h] = (_Float16)y;
    }
}

// ---------------------------------------------------------------------------
// Head: one wave32 per row.  q[row, nq] = feat[row,:]·Wh[:H] + qtau[nq] + bh
// ---------------------------------------------------------------------------
__global__ __launch_bounds__(256) void
head_kernel(const _Float16* __restrict__ feat, const float* __restrict__ Wh,
            const float* __restrict__ bh, const float* __restrict__ qtau,
            float* __restrict__ out)
{
    const int tid  = threadIdx.x;
    const int lane = tid & 31;
    const int wave = tid >> 5;
    const int row  = blockIdx.x * 8 + wave;

    const _Float16* f = feat + (size_t)row * kH + lane * 32;
    const float*    w = Wh + lane * 32;
    float s = 0.f;
#pragma unroll
    for (int i = 0; i < 4; ++i) {
        const v8h fv = *(const v8h*)(f + i * 8);
#pragma unroll
        for (int j = 0; j < 8; ++j) s += (float)fv[j] * w[i * 8 + j];
    }
#pragma unroll
    for (int off = 16; off > 0; off >>= 1) s += __shfl_xor(s, off, 32);

    const float base = s + bh[0];
    out[(size_t)row * kNQ + lane]      = base + qtau[lane];
    out[(size_t)row * kNQ + lane + 32] = base + qtau[lane + 32];
}

// ---------------------------------------------------------------------------
// Tau path (tiny, batch-independent):
//   v[j]   = sum_qe We2[qe,j] * Wh2[qe] ;  s = sum_qe be2[qe] * Wh2[qe]
//   qtau[n]= s + sum_j relu(tau_n*We1[j] + be1[j]) * v[j],  tau_n=(n+0.5)/64
// ---------------------------------------------------------------------------
__global__ void
tau_kernel(const float* __restrict__ We1, const float* __restrict__ be1,
           const float* __restrict__ We2, const float* __restrict__ be2,
           const float* __restrict__ Wh2, float* __restrict__ qtau)
{
    __shared__ float v[kQE];
    __shared__ float s_sh;
    const int j = threadIdx.x;
    float a = 0.f;
    for (int qe = 0; qe < kQE; ++qe) a += We2[qe * kQE + j] * Wh2[qe];
    v[j] = a;
    if (j == 0) {
        float s = 0.f;
        for (int qe = 0; qe < kQE; ++qe) s += be2[qe] * Wh2[qe];
        s_sh = s;
    }
    __syncthreads();
    const float tau = ((float)j + 0.5f) * (1.0f / kNQ);
    float q = s_sh;
    for (int jj = 0; jj < kQE; ++jj)
        q += fmaxf(tau * We1[jj] + be1[jj], 0.0f) * v[jj];
    qtau[j] = q;
}

// ---------------------------------------------------------------------------
// Launch: prep (f16 conversions, tau) once, then per critic:
//   gemm1 -> LN1 -> gemm2 -> LN2 -> head
// ---------------------------------------------------------------------------
extern "C" void kernel_launch(void* const* d_in, const int* in_sizes, int n_in,
                              void* d_out, int out_size, void* d_ws, size_t ws_size,
                              hipStream_t stream)
{
    const float* state  = (const float*)d_in[0];
    const float* action = (const float*)d_in[1];
    const float* We1    = (const float*)d_in[2];
    const float* be1    = (const float*)d_in[3];
    const float* We2    = (const float*)d_in[4];
    const float* be2    = (const float*)d_in[5];
    const float* Wf1    = (const float*)d_in[6];
    const float* bf1    = (const float*)d_in[7];
    const float* g1     = (const float*)d_in[8];
    const float* beta1  = (const float*)d_in[9];
    const float* Wf2    = (const float*)d_in[10];
    const float* bf2    = (const float*)d_in[11];
    const float* g2     = (const float*)d_in[12];
    const float* beta2  = (const float*)d_in[13];
    const float* Wh     = (const float*)d_in[14];
    const float* bh     = (const float*)d_in[15];
    float* out = (float*)d_out;

    // workspace layout
    char* ws = (char*)d_ws;
    float*    qtau  = (float*)ws;                                    //   512 B
    float*    Cbuf  = (float*)(ws + 1024);                           //  64 MB
    _Float16* act   = (_Float16*)(ws + 1024 + (size_t)kB*kH*4);      //  32 MB
    _Float16* xcat  = (_Float16*)((char*)act + (size_t)kB*kH*2);     // ~18 MB
    _Float16* Wf1h  = (_Float16*)((char*)xcat + (size_t)kB*kKIN*2);  // ~2.3 MB
    _Float16* Wf2h  = (_Float16*)((char*)Wf1h + (size_t)2*kH*kKIN*2);// ~4.2 MB

    // ---- prep: f16 conversions + tau path ----
    cvt_xcat_kernel<<<kB, 256, 0, stream>>>(state, action, xcat);
    {
        const int n4_1 = 2 * kH * kKIN / 4;
        cvt_f32_to_f16_kernel<<<(n4_1 + 255) / 256, 256, 0, stream>>>(Wf1, Wf1h, n4_1);
        const int n4_2 = 2 * kH * kH / 4;
        cvt_f32_to_f16_kernel<<<(n4_2 + 255) / 256, 256, 0, stream>>>(Wf2, Wf2h, n4_2);
    }
    for (int c = 0; c < 2; ++c)
        tau_kernel<<<1, kQE, 0, stream>>>(
            We1 + c * kQE, be1 + c * kQE, We2 + (size_t)c * kQE * kQE,
            be2 + c * kQE, Wh + (size_t)c * (kH + kQE) + kH, qtau + c * kNQ);

    const dim3 gg(kH / 128, kB / 256);   // 8 x 64 blocks

    for (int c = 0; c < 2; ++c) {
        gemm_async_kernel<kKIN><<<gg, 256, 0, stream>>>(
            xcat, Wf1h + (size_t)c * kH * kKIN, Cbuf);

        ln_relu_kernel<<<kB, 256, 0, stream>>>(
            Cbuf, bf1 + c * kH, g1 + c * kH, beta1 + c * kH, act);

        gemm_async_kernel<kH><<<gg, 256, 0, stream>>>(
            act, Wf2h + (size_t)c * kH * kH, Cbuf);

        ln_relu_kernel<<<kB, 256, 0, stream>>>(
            Cbuf, bf2 + c * kH, g2 + c * kH, beta2 + c * kH, act);

        head_kernel<<<kB / 8, 256, 0, stream>>>(
            act, Wh + (size_t)c * (kH + kQE), bh + c, qtau + c * kNQ,
            out + (size_t)c * kB * kNQ);
    }
}
